// SPRGNN_88648124991072
// MI455X (gfx1250) — matrix operand order repaired and checked by reading, not compiled, not run on hardware
//
#include <hip/hip_runtime.h>

typedef __attribute__((ext_vector_type(16))) _Float16 v16h;
typedef __attribute__((ext_vector_type(8)))  float    v8f;

#define N_NODES   100000
#define N_EDGES   1250000
#define N_GRAPHS  1024

// ---------------------------------------------------------------------------
// Zero-fill (d_ws is poisoned; zero accumulators every call, graph-safe)
// ---------------------------------------------------------------------------
__global__ void k_zero(float* __restrict__ p, long n) {
  long i = blockIdx.x * (long)blockDim.x + threadIdx.x;
  long stride = (long)gridDim.x * blockDim.x;
  for (; i < n; i += stride) p[i] = 0.0f;
}

// ---------------------------------------------------------------------------
// Embedding gather + lin (16 -> 32) + ReLU. 512 FMA/node, bandwidth-trivial.
// ---------------------------------------------------------------------------
__global__ void k_embed_lin(const int* __restrict__ tok,
                            const float* __restrict__ semb,
                            const float* __restrict__ cemb,
                            const float* __restrict__ w,   // [32,16]
                            const float* __restrict__ b,   // [32]
                            float* __restrict__ out) {     // [N,32]
  __shared__ float sw[32 * 16];
  __shared__ float sb[32];
  for (int i = threadIdx.x; i < 512; i += blockDim.x) sw[i] = w[i];
  if (threadIdx.x < 32) sb[threadIdx.x] = b[threadIdx.x];
  __syncthreads();

  int n = blockIdx.x * blockDim.x + threadIdx.x;
  if (n >= N_NODES) return;
  int s = tok[2 * n + 0];
  int c = tok[2 * n + 1];
  float f[16];
#pragma unroll
  for (int i = 0; i < 8; ++i) {
    f[i]     = semb[s * 8 + i];
    f[8 + i] = cemb[c * 8 + i];
  }
#pragma unroll
  for (int o = 0; o < 32; ++o) {
    float acc = sb[o];
#pragma unroll
    for (int k = 0; k < 16; ++k) acc = fmaf(f[k], sw[o * 16 + k], acc);
    out[(long)n * 32 + o] = fmaxf(acc, 0.0f);
  }
}

// ---------------------------------------------------------------------------
// Edge scatter-add: agg[dst] += x[src]. One thread per (edge, 4 channels).
// float4 coalesced gather + 4x global_atomic_add_f32. This is the roofline
// bottleneck of the whole net (~1 GB irregular traffic).
// ---------------------------------------------------------------------------
template <int K>
__global__ void k_scatter(const float* __restrict__ x,
                          const int* __restrict__ ei,   // [2, N_EDGES]
                          float* __restrict__ agg) {
  const int CH = K / 4;
  long t = blockIdx.x * (long)blockDim.x + threadIdx.x;
  if (t >= (long)N_EDGES * CH) return;
  int e = (int)(t / CH);
  int c = (int)(t % CH) * 4;
  int src = ei[e];
  int dst = ei[N_EDGES + e];
  const float4 v = *(const float4*)(x + (long)src * K + c);
  float* d = agg + (long)dst * K + c;
  atomicAdd(d + 0, v.x);
  atomicAdd(d + 1, v.y);
  atomicAdd(d + 2, v.z);
  atomicAdd(d + 3, v.w);
}

// ---------------------------------------------------------------------------
// GraphConv dense transform via WMMA:
//   out = relu( agg @ Wrel^T + x @ Wroot^T + b )   (both [N,K] x [K,64])
// One wave per 16x16 output tile; both GEMM terms fused into one f32
// accumulator chain (concatenated-K). N_NODES = 16*6250 -> no M tail.
// A fragment: ISA 16-bit A 16x32 layout, k = 2*(d&3) + 8*half + 16*(d>>2).
// B fragment: same k order as A, k = 2*(d&3) + 8*half + 16*(d>>2).
// C/D: VGPR r <-> row M = r + 8*half, col N = lane&15.
// ---------------------------------------------------------------------------
template <int K>
__global__ void k_graphconv_wmma(const float* __restrict__ agg,
                                 const float* __restrict__ x,
                                 const float* __restrict__ wrel,   // [64,K]
                                 const float* __restrict__ wroot,  // [64,K]
                                 const float* __restrict__ bias,   // [64]
                                 float* __restrict__ out) {        // [N,64]
  const int lane = threadIdx.x & 31;
  const int wave = threadIdx.x >> 5;
  const int half = lane >> 4;
  const int mr   = lane & 15;

  const int tile  = blockIdx.x * 8 + wave;  // 8 waves/block
  const int mt    = tile >> 2;              // node tile  (16 rows)
  const int ot    = tile & 3;               // out-ch tile (16 cols of 64)
  const int node0 = mt * 16;
  const int ocol  = ot * 16 + mr;           // this lane's output column

  v8f acc;
  const float bv = bias[ocol];
#pragma unroll
  for (int r = 0; r < 8; ++r) acc[r] = bv;

  const float* S[2] = {agg, x};
  const float* W[2] = {wrel, wroot};

#pragma unroll
  for (int s = 0; s < 2; ++s) {
    const float* Sp = S[s];
    const float* Wp = W[s];
#pragma unroll
    for (int kb = 0; kb < K; kb += 32) {
      v16h a, b;
#pragma unroll
      for (int d = 0; d < 8; ++d) {
        const int ka = kb + 2 * (d & 3) + 8 * half + 16 * (d >> 2);
        const float2 av = *(const float2*)(Sp + (long)(node0 + mr) * K + ka);
        a[2 * d + 0] = (_Float16)av.x;
        a[2 * d + 1] = (_Float16)av.y;
        const int kbb = kb + 2 * (d & 3) + 8 * half + 16 * (d >> 2);
        const float2 wv = *(const float2*)(Wp + (long)ocol * K + kbb);
        b[2 * d + 0] = (_Float16)wv.x;
        b[2 * d + 1] = (_Float16)wv.y;
      }
      acc = __builtin_amdgcn_wmma_f32_16x16x32_f16(
          /*neg_a=*/false, a, /*neg_b=*/false, b,
          /*c_mod=*/(short)0, acc, /*reuse_a=*/false, /*reuse_b=*/false);
    }
  }

#pragma unroll
  for (int r = 0; r < 8; ++r) {
    const int node = node0 + 8 * half + r;
    out[(long)node * 64 + ocol] = fmaxf(acc[r], 0.0f);
  }
}

// ---------------------------------------------------------------------------
// Global mean-pool: atomically accumulate per-graph sums and counts.
// ---------------------------------------------------------------------------
__global__ void k_pool(const float* __restrict__ h,      // [N,64]
                       const int* __restrict__ batch,
                       float* __restrict__ sums,         // [G,64]
                       float* __restrict__ counts) {     // [G]
  long t = blockIdx.x * (long)blockDim.x + threadIdx.x;
  if (t >= (long)N_NODES * 16) return;
  int n = (int)(t >> 4);
  int c = (int)(t & 15) * 4;
  int g = batch[n];
  const float4 v = *(const float4*)(h + (long)n * 64 + c);
  float* d = sums + (long)g * 64 + c;
  atomicAdd(d + 0, v.x);
  atomicAdd(d + 1, v.y);
  atomicAdd(d + 2, v.z);
  atomicAdd(d + 3, v.w);
  if ((t & 15) == 0) atomicAdd(counts + g, 1.0f);
}

// ---------------------------------------------------------------------------
// Classifier: [G,64] -> [G,10]. Tiny; one thread per output element.
// ---------------------------------------------------------------------------
__global__ void k_classify(const float* __restrict__ sums,
                           const float* __restrict__ counts,
                           const float* __restrict__ w,   // [10,64]
                           const float* __restrict__ b,   // [10]
                           float* __restrict__ out) {     // [G,10]
  int t = blockIdx.x * blockDim.x + threadIdx.x;
  if (t >= N_GRAPHS * 10) return;
  int g = t / 10;
  int c = t % 10;
  float inv = 1.0f / fmaxf(counts[g], 1.0f);
  float acc = b[c];
  const float* sg = sums + (long)g * 64;
  const float* wc = w + c * 64;
#pragma unroll
  for (int k = 0; k < 64; ++k) acc = fmaf(sg[k] * inv, wc[k], acc);
  out[t] = acc;
}

// ---------------------------------------------------------------------------
extern "C" void kernel_launch(void* const* d_in, const int* in_sizes, int n_in,
                              void* d_out, int out_size, void* d_ws, size_t ws_size,
                              hipStream_t stream) {
  const int*   tok   = (const int*)d_in[0];
  const int*   ei    = (const int*)d_in[1];
  const int*   batch = (const int*)d_in[2];
  const float* semb  = (const float*)d_in[3];
  const float* cemb  = (const float*)d_in[4];
  const float* lin_w = (const float*)d_in[5];
  const float* lin_b = (const float*)d_in[6];
  const float* w1r   = (const float*)d_in[7];
  const float* b1    = (const float*)d_in[8];
  const float* w1o   = (const float*)d_in[9];
  const float* w2r   = (const float*)d_in[10];
  const float* b2    = (const float*)d_in[11];
  const float* w2o   = (const float*)d_in[12];
  const float* cw    = (const float*)d_in[13];
  const float* cb    = (const float*)d_in[14];
  float* out = (float*)d_out;

  // Workspace layout (floats): ~90 MB total
  float* ws   = (float*)d_ws;
  float* x32  = ws;                              // N*32
  float* aggb = x32  + (long)N_NODES * 32;       // N*64 (conv1 uses first N*32)
  float* h1   = aggb + (long)N_NODES * 64;       // N*64
  float* h2   = h1   + (long)N_NODES * 64;       // N*64
  float* sums = h2   + (long)N_NODES * 64;       // G*64
  float* cnts = sums + (long)N_GRAPHS * 64;      // G

  const int B = 256;

  // 1. embed + lin + relu -> x32
  k_embed_lin<<<(N_NODES + B - 1) / B, B, 0, stream>>>(tok, semb, cemb, lin_w, lin_b, x32);

  // 2. conv1: zero agg, scatter-add, WMMA transform -> h1
  k_zero<<<2048, B, 0, stream>>>(aggb, (long)N_NODES * 32);
  {
    long tot = (long)N_EDGES * 8;
    k_scatter<32><<<(int)((tot + B - 1) / B), B, 0, stream>>>(x32, ei, aggb);
  }
  k_graphconv_wmma<32><<<(N_NODES / 16) * 4 / 8, B, 0, stream>>>(aggb, x32, w1r, w1o, b1, h1);

  // 3. conv2: zero agg, scatter-add, WMMA transform -> h2
  k_zero<<<2048, B, 0, stream>>>(aggb, (long)N_NODES * 64);
  {
    long tot = (long)N_EDGES * 16;
    k_scatter<64><<<(int)((tot + B - 1) / B), B, 0, stream>>>(h1, ei, aggb);
  }
  k_graphconv_wmma<64><<<(N_NODES / 16) * 4 / 8, B, 0, stream>>>(aggb, h1, w2r, w2o, b2, h2);

  // 4. mean pool
  k_zero<<<64, B, 0, stream>>>(sums, (long)N_GRAPHS * 64 + N_GRAPHS);
  {
    long tot = (long)N_NODES * 16;
    k_pool<<<(int)((tot + B - 1) / B), B, 0, stream>>>(h2, batch, sums, cnts);
  }

  // 5. classifier
  k_classify<<<(N_GRAPHS * 10 + B - 1) / B, B, 0, stream>>>(sums, cnts, cw, cb, out);
}
